// single_MogLSTM_24034636988674
// MI455X (gfx1250) — matrix operand order repaired and checked by reading, not compiled, not run on hardware
//
#include <hip/hip_runtime.h>
#include <hip/hip_bf16.h>

// ---------------------------------------------------------------------------
// Mogrifier LSTM, persistent-kernel formulation for MI455X (gfx1250, wave32).
//
// B=256,L=256,C=H=512. Recurrence is independent per batch row, so:
//   grid = 16 workgroups, each owns 16 batch rows for the whole sequence.
//   block = 512 threads = 16 waves; wave w owns H-columns [32w,32w+32)
//   (two 16x16 WMMA tiles) of every per-step intermediate, including the
//   persistent cell state Ct (v8f registers).
// All GEMMs use v_wmma_f32_16x16x32_bf16 (f32 accum). Weights are converted
// once (prep kernel) to transposed bf16 [N][K] in d_ws (6 MB) so B-fragments
// are single contiguous 32B loads per lane, streamed from L2 every step.
//
// NOTE: phase barriers carry an opaque asm memory clobber so LICM/CSE cannot
// hoist the (loop-invariant) weight-fragment loads across timesteps — that
// hoisting spilled ~1.4KB/lane to scratch in the previous revision and turned
// the WMMA B-operand stream into scratch reloads.
// ---------------------------------------------------------------------------

typedef __attribute__((ext_vector_type(16))) __bf16 v16bf;
typedef __attribute__((ext_vector_type(8)))  __bf16 v8bf;
typedef __attribute__((ext_vector_type(8)))  float  v8f;

#define B_  256
#define L_  256
#define C_  512
#define H_  512
#define K_  512          // every GEMM contraction dim is 512
#define BT  16           // batch rows per workgroup
#define NW  16           // waves per workgroup
#define THREADS (NW * 32)

__device__ __forceinline__ float sigm(float z) {
  return 1.0f / (1.0f + __expf(-z));
}

// barrier + opaque memory clobber: orders LDS phases AND blocks LICM/CSE of
// global weight loads across phases / timesteps (prevents scratch spilling).
__device__ __forceinline__ void phase_sync() {
  __syncthreads();
  asm volatile("" ::: "memory");
}

__device__ __forceinline__ v8f wmma_bf16(v16bf a, v16bf b, v8f c) {
  // (neg_a, A, neg_b, B, c_mod, C, reuse_a, reuse_b)
  return __builtin_amdgcn_wmma_f32_16x16x32_bf16(false, a, false, b,
                                                 (short)0, c, false, false);
}

// A fragment (16x32 bf16) from row-major [16][512] LDS tile.
// A layout (matches B): lane<16 -> M=lane, values K = hi*16+[0..15]
// within the 32-wide k-block (hi = lane>>4).
__device__ __forceinline__ v16bf ldfragA(const __bf16* S, int halflane, int hi,
                                         int kb) {
  const __bf16* base = S + halflane * K_ + kb * 32 + hi * 16;
  v8bf lo = *(const v8bf*)(base);
  v8bf hh = *(const v8bf*)(base + 8);
  return __builtin_shufflevector(lo, hh, 0, 1, 2, 3, 4, 5, 6, 7,
                                 8, 9, 10, 11, 12, 13, 14, 15);
}

// B fragment (32x16 bf16) from transposed weight Wt[N][K] (row-major, K=512).
// lanes 0-15: col = lane, K = kb*32 + [0..15]; lanes 16-31: K = kb*32+[16..31].
__device__ __forceinline__ v16bf ldfragB(const __bf16* Wt, int n, int halflane,
                                         int hi, int kb) {
  return *(const v16bf*)(Wt + (size_t)(n + halflane) * K_ + kb * 32 + hi * 16);
}

// acc{0,1} += A(16x512) @ Wt[:, n0..n0+31]  (two 16x16 tiles)
__device__ __forceinline__ void gemm2(const __bf16* A, const __bf16* Wt, int n0,
                                      v8f& acc0, v8f& acc1, int halflane,
                                      int hi) {
#pragma unroll
  for (int kb = 0; kb < 16; ++kb) {
    v16bf a = ldfragA(A, halflane, hi, kb);
    acc0 = wmma_bf16(a, ldfragB(Wt, n0, halflane, hi, kb), acc0);
    acc1 = wmma_bf16(a, ldfragB(Wt, n0 + 16, halflane, hi, kb), acc1);
  }
}

// acc[g*2+s] += A(16x512) @ Wt[:, g*512 + col0 + s*16 ...]   (8 tiles, 4 gates)
__device__ __forceinline__ void gemm8(const __bf16* A, const __bf16* Wt,
                                      int col0, v8f acc[8], int halflane,
                                      int hi) {
#pragma unroll
  for (int kb = 0; kb < 16; ++kb) {
    v16bf a = ldfragA(A, halflane, hi, kb);
#pragma unroll
    for (int g = 0; g < 4; ++g) {
#pragma unroll
      for (int s = 0; s < 2; ++s) {
        acc[g * 2 + s] = wmma_bf16(
            a, ldfragB(Wt, g * H_ + col0 + s * 16, halflane, hi, kb),
            acc[g * 2 + s]);
      }
    }
  }
}

// --------------------------- prep: W[K][N] f32 -> Wt[N][K] bf16 -------------
__global__ void transpose_to_bf16(const float* __restrict__ src,
                                  __bf16* __restrict__ dst, int K, int N) {
  int idx = blockIdx.x * blockDim.x + threadIdx.x;
  if (idx >= K * N) return;
  int n = idx / K;
  int k = idx - n * K;
  dst[idx] = (__bf16)src[k * N + n];
}

// --------------------------- main persistent kernel -------------------------
__global__ __launch_bounds__(THREADS, 1) void moglstm_kernel(
    const float* __restrict__ x, const __bf16* __restrict__ Qt,
    const __bf16* __restrict__ Rt, const __bf16* __restrict__ Wmxt,
    const __bf16* __restrict__ Wmht, const __bf16* __restrict__ Wiht,
    const __bf16* __restrict__ Whmt, const float* __restrict__ bmx,
    const float* __restrict__ bmh, const float* __restrict__ bih,
    const float* __restrict__ bhm, float* __restrict__ out) {
  __shared__ __align__(32) __bf16 h_s[BT * K_];
  __shared__ __align__(32) __bf16 x_s[BT * K_];
  __shared__ __align__(32) __bf16 m_s[BT * K_];

  const int tid = threadIdx.x;
  const int w = tid >> 5;        // wave id 0..15
  const int lane = tid & 31;
  const int halflane = lane & 15;
  const int hi = lane >> 4;      // 0/1 half of the wave
  const int col0 = w * 32;       // this wave's H-column base (2 strips)
  const int rb = blockIdx.x * BT;

  const int c0 = col0 + halflane;   // column of strip 0 for this lane
  const int c1 = c0 + 16;           // column of strip 1

  // hoisted (time-invariant) bias sums — live in SGPR/VGPR, immune to clobber
  const float bm0 = bmx[c0] + bmh[c0];
  const float bm1 = bmx[c1] + bmh[c1];
  float bg[4][2];
#pragma unroll
  for (int g = 0; g < 4; ++g) {
    bg[g][0] = bih[g * H_ + c0] + bhm[g * H_ + c0];
    bg[g][1] = bih[g * H_ + c1] + bhm[g * H_ + c1];
  }

  // h0 = 0
  for (int i = tid; i < BT * K_; i += THREADS) h_s[i] = (__bf16)0.0f;

  // persistent cell state: rows hi*8..hi*8+7 at columns c0 / c1
  v8f Ct0 = {};
  v8f Ct1 = {};

  phase_sync();

  for (int t = 0; t < L_; ++t) {
    // ---- load x[rb..rb+15, t, :] tile, f32 -> bf16 into LDS --------------
    {
      const float* src = x + ((size_t)(rb + w) * L_ + t) * C_ + lane * 16;
      __bf16* dst = x_s + w * K_ + lane * 16;
#pragma unroll
      for (int j = 0; j < 16; ++j) dst[j] = (__bf16)src[j];
      if (t + 1 < L_) {
        // pull next timestep's x row segment toward L2/L0 (global_prefetch_b8)
        __builtin_prefetch(src + C_, 0, 1);
      }
    }
    phase_sync();

    // ---- Mogrify: it=0,2,4 -> x = 2*sig(h@Q)*x ; it=1,3 -> h = 2*sig(x@R)*h
    // kept as a real loop (no unroll) so Qt/Rt fragments are not CSE'd into
    // a register set that cannot fit (previously spilled to scratch).
#pragma unroll 1
    for (int itr = 0; itr < 5; ++itr) {
      const bool updX = ((itr & 1) == 0);
      const __bf16* A = updX ? h_s : x_s;
      const __bf16* Wt = updX ? Qt : Rt;
      __bf16* Tgt = updX ? x_s : h_s;
      v8f a0 = {};
      v8f a1 = {};
      gemm2(A, Wt, col0, a0, a1, halflane, hi);
#pragma unroll
      for (int r = 0; r < 8; ++r) {
        const int row = hi * 8 + r;
        const float g0 = 2.0f * sigm(a0[r]);
        const float g1 = 2.0f * sigm(a1[r]);
        const float v0 = (float)Tgt[row * K_ + c0];
        const float v1 = (float)Tgt[row * K_ + c1];
        Tgt[row * K_ + c0] = (__bf16)(g0 * v0);
        Tgt[row * K_ + c1] = (__bf16)(g1 * v1);
      }
      phase_sync();
    }

    // ---- mt = x@Wmx + h@Wmh + bias ---------------------------------------
    {
      v8f a0 = {};
      v8f a1 = {};
      gemm2(x_s, Wmxt, col0, a0, a1, halflane, hi);
      gemm2(h_s, Wmht, col0, a0, a1, halflane, hi);
#pragma unroll
      for (int r = 0; r < 8; ++r) {
        const int row = hi * 8 + r;
        m_s[row * K_ + c0] = (__bf16)(a0[r] + bm0);
        m_s[row * K_ + c1] = (__bf16)(a1[r] + bm1);
      }
    }
    phase_sync();

    // ---- gates = x@Wih + m@Whm + bias  (4 gates x 2 strips) --------------
    v8f acc[8];
    {
      const v8f z = {};
#pragma unroll
      for (int i = 0; i < 8; ++i) acc[i] = z;
    }
    gemm8(x_s, Wiht, col0, acc, halflane, hi);
    gemm8(m_s, Whmt, col0, acc, halflane, hi);
    phase_sync();  // all x_s/m_s reads done before next-step overwrite

    // ---- LSTM elementwise; Ct persistent in registers --------------------
#pragma unroll
    for (int s = 0; s < 2; ++s) {
      v8f& Cs = s ? Ct1 : Ct0;
      const int cc = s ? c1 : c0;
#pragma unroll
      for (int r = 0; r < 8; ++r) {
        const int row = hi * 8 + r;
        const float ig = sigm(acc[0 * 2 + s][r] + bg[0][s]);
        const float fg = sigm(acc[1 * 2 + s][r] + bg[1][s]);
        const float cg = tanhf(acc[2 * 2 + s][r] + bg[2][s]);
        const float og = sigm(acc[3 * 2 + s][r] + bg[3][s]);
        const float c = fg * Cs[r] + ig * cg;
        Cs[r] = c;
        const float h = og * tanhf(c);
        h_s[row * K_ + cc] = (__bf16)h;
        out[((size_t)(rb + row) * L_ + t) * H_ + cc] = h;
      }
    }
    // next-iteration top barrier (after x tile store) orders h_s visibility
  }
}

// ---------------------------------------------------------------------------
extern "C" void kernel_launch(void* const* d_in, const int* in_sizes, int n_in,
                              void* d_out, int out_size, void* d_ws,
                              size_t ws_size, hipStream_t stream) {
  (void)in_sizes; (void)n_in; (void)out_size; (void)ws_size;

  const float* x     = (const float*)d_in[0];
  const float* Wih_w = (const float*)d_in[1];
  const float* Wih_b = (const float*)d_in[2];
  const float* Wmx_w = (const float*)d_in[3];
  const float* Wmx_b = (const float*)d_in[4];
  const float* Wmh_w = (const float*)d_in[5];
  const float* Wmh_b = (const float*)d_in[6];
  const float* Whm_w = (const float*)d_in[7];
  const float* Whm_b = (const float*)d_in[8];
  const float* Q     = (const float*)d_in[9];
  const float* R     = (const float*)d_in[10];

  // bf16 transposed weights in workspace (6 MB total)
  __bf16* ws   = (__bf16*)d_ws;
  __bf16* Qt   = ws;                 // 512x512
  __bf16* Rt   = ws + 262144;        // 512x512
  __bf16* Wmxt = ws + 524288;        // 512x512
  __bf16* Wmht = ws + 786432;        // 512x512
  __bf16* Wiht = ws + 1048576;       // 2048x512
  __bf16* Whmt = ws + 2097152;       // 2048x512

  const int tb = 256;
  transpose_to_bf16<<<(512 * 512 + tb - 1) / tb, tb, 0, stream>>>(Q, Qt, 512, 512);
  transpose_to_bf16<<<(512 * 512 + tb - 1) / tb, tb, 0, stream>>>(R, Rt, 512, 512);
  transpose_to_bf16<<<(512 * 512 + tb - 1) / tb, tb, 0, stream>>>(Wmx_w, Wmxt, 512, 512);
  transpose_to_bf16<<<(512 * 512 + tb - 1) / tb, tb, 0, stream>>>(Wmh_w, Wmht, 512, 512);
  transpose_to_bf16<<<(512 * 2048 + tb - 1) / tb, tb, 0, stream>>>(Wih_w, Wiht, 512, 2048);
  transpose_to_bf16<<<(512 * 2048 + tb - 1) / tb, tb, 0, stream>>>(Whm_w, Whmt, 512, 2048);

  moglstm_kernel<<<B_ / BT, THREADS, 0, stream>>>(
      x, Qt, Rt, Wmxt, Wmht, Wiht, Whmt, Wmx_b, Wmh_b, Wih_b, Whm_b,
      (float*)d_out);
}
